// MultiHeadSelfAttention_30906584662370
// MI455X (gfx1250) — matrix operand order repaired and checked by reading, not compiled, not run on hardware
//
#include <hip/hip_runtime.h>

// ---------------------------------------------------------------------------
// Problem constants (from reference)
// ---------------------------------------------------------------------------
#define B_   2
#define S_   2048
#define DM   1024
#define HQ   16
#define HKV  4
#define DH   64
#define MROWS (B_ * S_)        // 4096 token rows
#define NKV   (HKV * DH)       // 256

typedef __attribute__((ext_vector_type(16))) __bf16 v16bf;
typedef __attribute__((ext_vector_type(8)))  float  v8f;

typedef unsigned int u32x4 __attribute__((ext_vector_type(4)));
typedef int          i32x8 __attribute__((ext_vector_type(8)));
typedef int          i32x4 __attribute__((ext_vector_type(4)));

union FragU { v16bf v; uint4 q[2]; };

// ---------------------------------------------------------------------------
// bf16 helpers (round-to-nearest-even)
// ---------------------------------------------------------------------------
static __device__ __forceinline__ unsigned short f2bf_bits(float f) {
  unsigned u = __builtin_bit_cast(unsigned, f);
  u += 0x7FFFu + ((u >> 16) & 1u);
  return (unsigned short)(u >> 16);
}
static __device__ __forceinline__ __bf16 f2bf(float f) {
  unsigned short h = f2bf_bits(f);
  return __builtin_bit_cast(__bf16, h);
}
static __device__ __forceinline__ float bf2f(__bf16 h) {
  unsigned u = ((unsigned)__builtin_bit_cast(unsigned short, h)) << 16;
  return __builtin_bit_cast(float, u);
}

static __device__ __forceinline__ v8f wmma_bf16(v16bf a, v16bf b, v8f c) {
  // v_wmma_f32_16x16x32_bf16 : D(16x16 f32) = A(16x32) * B(32x16) + C
  return __builtin_amdgcn_wmma_f32_16x16x32_bf16(false, a, false, b,
                                                 (short)0, c, false, false);
}

// ---------------------------------------------------------------------------
// Tensor Data Mover: async DMA of a 2-D bf16 tile (tile_d0 contiguous elems x
// tile_d1 rows, row stride `stride0` elems) from global into LDS.
// D# bitfields per CDNA5 ISA (data_size=1 -> 2 bytes,
// count=1, type=2 "image", no padding / iterate / gather / multicast).
// Issue once per workgroup (wave 0); completion tracked on TENSORcnt.
// ---------------------------------------------------------------------------
static __device__ __forceinline__ unsigned lds_off(const void* p) {
  // generic LDS address: low 32 bits are the LDS offset (ISA 10.2 aperture rules)
  return (unsigned)(size_t)p;
}

static __device__ __forceinline__ void tdm_load_2d(const void* gsrc, unsigned ldsaddr,
                                                   unsigned tile_d0, unsigned tile_d1,
                                                   unsigned tensor_d0, unsigned tensor_d1,
                                                   unsigned stride0_elems) {
  unsigned long long ga = (unsigned long long)(size_t)gsrc;
  u32x4 g0;
  g0[0] = 1u;                                           // count=1 (valid descriptor)
  g0[1] = ldsaddr;                                      // lds_addr (bytes)
  g0[2] = (unsigned)(ga & 0xFFFFFFFFu);                 // global_addr[31:0]
  g0[3] = (unsigned)((ga >> 32) & 0x01FFFFFFu)          // global_addr[56:32]
        | (2u << 30);                                   // type = 2
  i32x8 g1;
  g1[0] = (int)(1u << 16);                              // mask=0, data_size=1 (2B)
  g1[1] = (int)((tensor_d0 & 0xFFFFu) << 16);           // tensor_dim0[15:0]
  g1[2] = (int)(((tensor_d0 >> 16) & 0xFFFFu) |         // tensor_dim0[31:16]
                ((tensor_d1 & 0xFFFFu) << 16));         // tensor_dim1[15:0]
  g1[3] = (int)(((tensor_d1 >> 16) & 0xFFFFu) |         // tensor_dim1[31:16]
                ((tile_d0 & 0xFFFFu) << 16));           // tile_dim0
  g1[4] = (int)(tile_d1 & 0xFFFFu);                     // tile_dim1 (tile_dim2=0)
  g1[5] = (int)stride0_elems;                           // tensor_dim0_stride[31:0]
  g1[6] = 0;                                            // stride0[47:32] | stride1[15:0]
  g1[7] = 0;                                            // stride1[47:16]
  i32x4 z4 = {0, 0, 0, 0};
#if __clang_major__ >= 23
  i32x8 z8 = {0, 0, 0, 0, 0, 0, 0, 0};
  __builtin_amdgcn_tensor_load_to_lds(g0, g1, z4, z4, z8, 0);
#else
  __builtin_amdgcn_tensor_load_to_lds(g0, g1, z4, z4, 0);
#endif
}

// ---------------------------------------------------------------------------
// 1) fp32 -> bf16 conversion (4-wide)
// ---------------------------------------------------------------------------
__global__ void k_cvt(const float* __restrict__ s, __bf16* __restrict__ d, int n4) {
  int i = blockIdx.x * blockDim.x + threadIdx.x;
  if (i >= n4) return;
  float4 v = ((const float4*)s)[i];
  ushort4 o;
  o.x = f2bf_bits(v.x); o.y = f2bf_bits(v.y);
  o.z = f2bf_bits(v.z); o.w = f2bf_bits(v.w);
  ((ushort4*)d)[i] = o;
}

// ---------------------------------------------------------------------------
// 2) bf16 GEMM:  OUT[M,N] = A[M,K] * W[N,K]^T   (128x128 block, 8 waves)
//    TDM double-buffered K-tiles; wave grid 4(M) x 2(N); 16x16x32 WMMA.
// ---------------------------------------------------------------------------
template <int OUT_BF16>
__global__ __launch_bounds__(256) void k_gemm(const __bf16* __restrict__ A,
                                              const __bf16* __restrict__ W,
                                              float*  __restrict__ outF,
                                              __bf16* __restrict__ outB,
                                              int M, int N, int K) {
  __shared__ __align__(16) __bf16 As[2][128 * 32];   // 2 x 8 KB
  __shared__ __align__(16) __bf16 Ws[2][128 * 32];   // 2 x 8 KB

  const int t = threadIdx.x, lane = t & 31, w = t >> 5;
  const int wm = w >> 1, wn = w & 1;
  const int m0 = blockIdx.y * 128, n0 = blockIdx.x * 128;

  const v8f vzero = {0.f,0.f,0.f,0.f,0.f,0.f,0.f,0.f};
  v8f acc[2][4];
  for (int mi = 0; mi < 2; ++mi)
    for (int ni = 0; ni < 4; ++ni) acc[mi][ni] = vzero;

  const int fm = lane & 15;
  const int kA = (lane >> 4) * 8;    // A frag: K chunks {kA..kA+7, kA+16..kA+23}
  const int kB = (lane >> 4) * 8;    // B frag: K chunks {kB..kB+7, kB+16..kB+23}
  const int nsteps = K / 32;

  for (int ks = 0; ks < nsteps; ++ks) {
    __syncthreads();                 // prior reads of the buffer we re-fill are done
    if (w == 0) {
      if (ks == 0) {
        tdm_load_2d(A + (size_t)m0 * K, lds_off(As[0]), 32, 128, 32, 128, K);
        tdm_load_2d(W + (size_t)n0 * K, lds_off(Ws[0]), 32, 128, 32, 128, K);
      }
      if (ks + 1 < nsteps) {
        int k1 = (ks + 1) * 32, nb = (ks + 1) & 1;
        tdm_load_2d(A + (size_t)m0 * K + k1, lds_off(As[nb]), 32, 128, 32, 128, K);
        tdm_load_2d(W + (size_t)n0 * K + k1, lds_off(Ws[nb]), 32, 128, 32, 128, K);
        __builtin_amdgcn_s_wait_tensorcnt(2);   // oldest pair (current buf) landed
      } else {
        __builtin_amdgcn_s_wait_tensorcnt(0);
      }
    }
    __syncthreads();

    const __bf16* Asb = As[ks & 1];
    const __bf16* Wsb = Ws[ks & 1];
    v16bf af[2], bfr[4];
    for (int mi = 0; mi < 2; ++mi) {
      const __bf16* base = Asb + (wm * 32 + mi * 16 + fm) * 32;
      FragU u;
      u.q[0] = *(const uint4*)(base + kA);
      u.q[1] = *(const uint4*)(base + kA + 16);
      af[mi] = u.v;
    }
    for (int ni = 0; ni < 4; ++ni) {
      const __bf16* base = Wsb + (wn * 64 + ni * 16 + fm) * 32;
      FragU u;
      u.q[0] = *(const uint4*)(base + kB);
      u.q[1] = *(const uint4*)(base + kB + 16);
      bfr[ni] = u.v;
    }
    for (int mi = 0; mi < 2; ++mi)
      for (int ni = 0; ni < 4; ++ni)
        acc[mi][ni] = wmma_bf16(af[mi], bfr[ni], acc[mi][ni]);
  }

  // C layout: VGPR r, lanes 0-15 -> M=r, lanes 16-31 -> M=r+8; N = lane&15
  const int hh = (lane >> 4) * 8;
  for (int mi = 0; mi < 2; ++mi)
    for (int ni = 0; ni < 4; ++ni) {
      int gn  = n0 + wn * 64 + ni * 16 + fm;
      int gm0 = m0 + wm * 32 + mi * 16 + hh;
      for (int r = 0; r < 8; ++r) {
        size_t off = (size_t)(gm0 + r) * N + gn;
        if (OUT_BF16) outB[off] = f2bf(acc[mi][ni][r]);
        else          outF[off] = acc[mi][ni][r];
      }
    }
}

// ---------------------------------------------------------------------------
// 3) RoPE in-place on bf16 Q/K. block = nHeads*32 (one pair per thread)
// ---------------------------------------------------------------------------
__global__ void k_rope(__bf16* __restrict__ qk, int nHeads) {
  int row = blockIdx.x;
  int s   = row & (S_ - 1);
  int h   = threadIdx.x >> 5;
  int i   = threadIdx.x & 31;
  float inv = __powf(10000.0f, -(float)i * (1.0f / 32.0f));
  float ang = (float)s * inv;
  float sn, cs;
  __sincosf(ang, &sn, &cs);
  __bf16* p = qk + ((size_t)row * nHeads + h) * DH + 2 * i;
  float x1 = bf2f(p[0]), x2 = bf2f(p[1]);
  p[0] = f2bf(x1 * cs - x2 * sn);
  p[1] = f2bf(x1 * sn + x2 * cs);
}

// ---------------------------------------------------------------------------
// 4) V transpose: [b,s,hk,d] -> [b,hk,d,s]  (so B-fragments of V are contiguous)
// ---------------------------------------------------------------------------
__global__ void k_tv(const __bf16* __restrict__ V, __bf16* __restrict__ Vt) {
  int idx = blockIdx.x * 256 + threadIdx.x;       // over 2^20 elements
  int d  = idx & 63;
  int hk = (idx >> 6) & 3;
  int s  = (idx >> 8) & (S_ - 1);
  int b  = idx >> 19;
  Vt[(((size_t)b * HKV + hk) * DH + d) * S_ + s] = V[idx];
}

// ---------------------------------------------------------------------------
// 5) Flash attention (causal, GQA). block = 256 (8 waves), each wave = 16 Q rows.
//    TDM double-buffered K / V^T tiles; DMA of tile kb+1 overlaps softmax+WMMA
//    of tile kb.
//    Q: [b,s,16,64] bf16 (RoPE'd)   K: [b,s,4,64] bf16 (RoPE'd)
//    Vt: [b,4,64,s] bf16            O: [b,s,16,64] bf16
// ---------------------------------------------------------------------------
__global__ __launch_bounds__(256) void k_flash(const __bf16* __restrict__ Q,
                                               const __bf16* __restrict__ Kg,
                                               const __bf16* __restrict__ Vt,
                                               __bf16* __restrict__ O) {
  __shared__ __align__(16) __bf16 Ks[2][128 * 64];   // K tiles   (kv x d)  2x16 KB
  __shared__ __align__(16) __bf16 Vs[2][64 * 128];   // V^T tiles (d x kv)  2x16 KB
  __shared__ __align__(16) __bf16 Ps[8 * 16 * 128];  // per-wave P          32 KB

  const int qb = blockIdx.x, h = blockIdx.y, b = blockIdx.z;
  const int hk = h >> 2;
  const int t = threadIdx.x, lane = t & 31, w = t >> 5;
  const int q0 = qb * 128 + w * 16;
  const int fm = lane & 15;
  const int hh = (lane >> 4) * 8;      // C-layout row offset
  const int kA = (lane >> 4) * 8;      // A-frag K select
  const int kB = (lane >> 4) * 8;      // B-frag K select

  // Q strip (16x64) as two 16x32 A-fragments, straight from global
  v16bf qa[2];
  {
    const __bf16* qrow = Q + (((size_t)b * S_ + q0 + fm) * HQ + h) * DH;
    FragU u0, u1;
    u0.q[0] = *(const uint4*)(qrow + kA);
    u0.q[1] = *(const uint4*)(qrow + kA + 16);
    u1.q[0] = *(const uint4*)(qrow + 32 + kA);
    u1.q[1] = *(const uint4*)(qrow + 32 + kA + 16);
    qa[0] = u0.v; qa[1] = u1.v;
  }

  const __bf16* kbase = Kg + ((size_t)b * S_ * HKV + hk) * DH;   // + kv*NKV
  const __bf16* vbase = Vt + ((size_t)b * HKV + hk) * DH * S_;   // + d*S_ + kv

  const v8f vzero = {0.f,0.f,0.f,0.f,0.f,0.f,0.f,0.f};
  float mrow[8], lrow[8];
  v8f o_acc[4];
  for (int r = 0; r < 8; ++r) { mrow[r] = -1e30f; lrow[r] = 0.f; }
  for (int nb = 0; nb < 4; ++nb) o_acc[nb] = vzero;

  for (int kb = 0; kb <= qb; ++kb) {
    const int cur = kb & 1;
    __syncthreads();                  // everyone done reading the buffer being refilled
    if (w == 0) {
      if (kb == 0) {
        tdm_load_2d(kbase, lds_off(Ks[0]), DH, 128, DH, 128, NKV);
        tdm_load_2d(vbase, lds_off(Vs[0]), 128, DH, 128, DH, S_);
      }
      if (kb + 1 <= qb) {
        int kn = kb + 1, nb2 = kn & 1;
        tdm_load_2d(kbase + (size_t)kn * 128 * NKV, lds_off(Ks[nb2]), DH, 128, DH, 128, NKV);
        tdm_load_2d(vbase + kn * 128,               lds_off(Vs[nb2]), 128, DH, 128, DH, S_);
        __builtin_amdgcn_s_wait_tensorcnt(2);   // current tile's pair has landed
      } else {
        __builtin_amdgcn_s_wait_tensorcnt(0);
      }
    }
    __syncthreads();
    const __bf16* Ksb = Ks[cur];
    const __bf16* Vsb = Vs[cur];

    // ---- S = Q @ K^T  (16 x 128) as 8 x (16x16) blocks, 2 WMMA each ----
    v8f sblk[8];
    for (int kt = 0; kt < 8; ++kt) {
      const __bf16* base = Ksb + (kt * 16 + fm) * 64;   // B column = K row
      FragU u0, u1;
      u0.q[0] = *(const uint4*)(base + kB);
      u0.q[1] = *(const uint4*)(base + kB + 16);
      u1.q[0] = *(const uint4*)(base + 32 + kB);
      u1.q[1] = *(const uint4*)(base + 32 + kB + 16);
      v8f z = vzero;
      z = wmma_bf16(qa[0], u0.v, z);
      z = wmma_bf16(qa[1], u1.v, z);
      sblk[kt] = z;
    }

    // ---- clamp(+/-80), scale 1/sqrt(64), causal mask on diagonal block ----
    const bool diag = (kb == qb);
    for (int kt = 0; kt < 8; ++kt)
      for (int r = 0; r < 8; ++r) {
        float v = sblk[kt][r];
        v = fminf(fmaxf(v, -80.f), 80.f) * 0.125f;
        if (diag) {
          int qi = q0 + r + hh;
          int ci = kb * 128 + kt * 16 + fm;
          if (ci > qi) v = -3.0e38f;
        }
        sblk[kt][r] = v;
      }

    // ---- online softmax (row stats live per lane-half, 16-lane reductions) ----
    float scl[8];
    for (int r = 0; r < 8; ++r) {
      float mx = sblk[0][r];
      for (int kt = 1; kt < 8; ++kt) mx = fmaxf(mx, sblk[kt][r]);
      for (int off = 8; off; off >>= 1) mx = fmaxf(mx, __shfl_xor(mx, off, 16));
      float mn = fmaxf(mrow[r], mx);
      scl[r] = __expf(mrow[r] - mn);
      float rs = 0.f;
      for (int kt = 0; kt < 8; ++kt) {
        float p = __expf(sblk[kt][r] - mn);
        sblk[kt][r] = p;
        rs += p;
      }
      for (int off = 8; off; off >>= 1) rs += __shfl_xor(rs, off, 16);
      lrow[r] = lrow[r] * scl[r] + rs;
      mrow[r] = mn;
    }
    for (int nb = 0; nb < 4; ++nb)
      for (int r = 0; r < 8; ++r) o_acc[nb][r] *= scl[r];

    // ---- P: C-layout -> LDS -> A-layout ----
    __bf16* pw = Ps + w * 16 * 128;
    for (int kt = 0; kt < 8; ++kt)
      for (int r = 0; r < 8; ++r)
        pw[(r + hh) * 128 + kt * 16 + fm] = f2bf(sblk[kt][r]);
    __syncthreads();

    v16bf pa[4];
    for (int kc = 0; kc < 4; ++kc) {
      const __bf16* base = pw + fm * 128 + kc * 32;
      FragU u;
      u.q[0] = *(const uint4*)(base + kA);
      u.q[1] = *(const uint4*)(base + kA + 16);
      pa[kc] = u.v;
    }

    // ---- O += P @ V  (B frags contiguous thanks to V^T tile) ----
    for (int nb = 0; nb < 4; ++nb)
      for (int kc = 0; kc < 4; ++kc) {
        const __bf16* base = Vsb + (nb * 16 + fm) * 128 + kc * 32 + kB;
        FragU u;
        u.q[0] = *(const uint4*)(base);
        u.q[1] = *(const uint4*)(base + 16);
        o_acc[nb] = wmma_bf16(pa[kc], u.v, o_acc[nb]);
      }
  }

  // ---- normalize & store attn output [b,s,h,d] as bf16 ----
  for (int nb = 0; nb < 4; ++nb)
    for (int r = 0; r < 8; ++r) {
      int qi = q0 + r + hh;
      int dc = nb * 16 + fm;
      float val = o_acc[nb][r] / lrow[r];
      O[(((size_t)b * S_ + qi) * HQ + h) * DH + dc] = f2bf(val);
    }
}

// ---------------------------------------------------------------------------
// Host-side orchestration
// ---------------------------------------------------------------------------
extern "C" void kernel_launch(void* const* d_in, const int* in_sizes, int n_in,
                              void* d_out, int out_size, void* d_ws, size_t ws_size,
                              hipStream_t stream) {
  (void)in_sizes; (void)n_in; (void)out_size; (void)ws_size;
  const float* x  = (const float*)d_in[0];
  const float* Wq = (const float*)d_in[1];
  const float* Wk = (const float*)d_in[2];
  const float* Wv = (const float*)d_in[3];
  const float* Wo = (const float*)d_in[4];
  float* out = (float*)d_out;

  char* p = (char*)d_ws;
  auto take = [&](size_t elems) {
    __bf16* r = (__bf16*)p;
    p += (elems * 2 + 255) & ~(size_t)255;
    return r;
  };
  __bf16* xb  = take((size_t)MROWS * DM);
  __bf16* wqb = take((size_t)DM * DM);
  __bf16* wkb = take((size_t)NKV * DM);
  __bf16* wvb = take((size_t)NKV * DM);
  __bf16* wob = take((size_t)DM * DM);
  __bf16* Qb  = take((size_t)MROWS * DM);
  __bf16* Kbf = take((size_t)MROWS * NKV);
  __bf16* Vb  = take((size_t)MROWS * NKV);
  __bf16* Vt  = take((size_t)MROWS * NKV);
  __bf16* A2  = take((size_t)MROWS * DM);

  auto cvt = [&](const float* s, __bf16* d, size_t n) {
    int n4 = (int)(n / 4);
    k_cvt<<<dim3((n4 + 255) / 256), dim3(256), 0, stream>>>(s, d, n4);
  };
  cvt(x,  xb,  (size_t)MROWS * DM);
  cvt(Wq, wqb, (size_t)DM * DM);
  cvt(Wk, wkb, (size_t)NKV * DM);
  cvt(Wv, wvb, (size_t)NKV * DM);
  cvt(Wo, wob, (size_t)DM * DM);

  dim3 gq(DM / 128, MROWS / 128);     // (8, 32)
  dim3 gk(NKV / 128, MROWS / 128);    // (2, 32)
  k_gemm<1><<<gq, 256, 0, stream>>>(xb, wqb, nullptr, Qb,  MROWS, DM,  DM);
  k_gemm<1><<<gk, 256, 0, stream>>>(xb, wkb, nullptr, Kbf, MROWS, NKV, DM);
  k_gemm<1><<<gk, 256, 0, stream>>>(xb, wvb, nullptr, Vb,  MROWS, NKV, DM);

  k_rope<<<dim3(MROWS), dim3(HQ * 32), 0, stream>>>(Qb, HQ);
  k_rope<<<dim3(MROWS), dim3(HKV * 32), 0, stream>>>(Kbf, HKV);

  k_tv<<<dim3((MROWS * NKV) / 256), dim3(256), 0, stream>>>(Vb, Vt);

  dim3 gf(S_ / 128, HQ, B_);
  k_flash<<<gf, dim3(256), 0, stream>>>(Qb, Kbf, Vt, A2);

  k_gemm<0><<<gq, 256, 0, stream>>>(A2, wob, out, nullptr, MROWS, DM, DM);
}
